// CRF_79439715107050
// MI455X (gfx1250) — matrix operand order repaired and checked by reading, not compiled, not run on hardware
//
#include <hip/hip_runtime.h>
#include <hip/hip_bf16.h>
#include <stdint.h>

typedef __attribute__((ext_vector_type(16))) _Float16 v16h;
typedef __attribute__((ext_vector_type(8)))  float    v8f;

#define B_SZ     512
#define S_LEN    2048
#define T_TAGS   32
#define N_CHUNK  16
#define CHUNK_LEN 128

union FragU { unsigned int u[8]; uint4 q[2]; v16h v; };

static __device__ __forceinline__ unsigned int pack_f16(float x, float y) {
  union { _Float16 h[2]; unsigned int u; } p;
  p.h[0] = (_Float16)x; p.h[1] = (_Float16)y;
  return p.u;
}

// Per-wave LDS scratch: e-vector + f16 staging tile for D->B relayout.
// Zh rows are 40 halfs (80B): 16B-aligned and bank-conflict-free across 16 lanes.
struct WaveScratch {
  float sE[32];
  unsigned short Zh[32][40];
};

// ---------------- Stage 0: E = exp(transitions) ----------------
__global__ void crf_prep_expT(const float* __restrict__ trans, float* __restrict__ Emat) {
  int i = threadIdx.x + blockIdx.x * blockDim.x;
  if (i < T_TAGS * T_TAGS) Emat[i] = __expf(trans[i]);
}

// ---------------- Stage 1: chunked matrix scan (WMMA) ----------------
// worker = b*16 + k computes M_k^T = prod_{t in chunk} diag(e_t) * E^T
// (applied left for ascending t), i.e. the transpose of the chunk operator
// M_k = prod(E*diag(e_t)).  State Z kept as two f16 B-fragments; A = E^T
// constant f16 fragments.  Renormalize by an exact power of two every 4 steps.
__global__ void __launch_bounds__(256)
crf_chunk_scan(const float* __restrict__ emis, const float* __restrict__ Emat,
               float* __restrict__ mats, int* __restrict__ exps)
{
  __shared__ WaveScratch scr[8];
  const int lane = threadIdx.x & 31;
  const int wid  = threadIdx.x >> 5;
  const int worker = blockIdx.x * 8 + wid;      // 0..8191
  const int b = worker >> 4;
  const int k = worker & 15;
  const int h  = lane >> 4;                     // lane half (K/M half select)
  const int ln = lane & 15;
  WaveScratch& S = scr[wid];

  // A operand: E^T as two constant f16 16x32 fragments.
  // A layout (contiguous k order, same as B): lane holds M=ln(+16r); VGPR v holds packed K pair
  // k0 = 2*v + 16*h.  A[m][k] = E[k][m].
  FragU af[2];
  #pragma unroll
  for (int r = 0; r < 2; ++r) {
    const int m = ln + 16 * r;
    #pragma unroll
    for (int v = 0; v < 8; ++v) {
      const int k0 = 2 * v + 16 * h;
      af[r].u[v] = pack_f16(Emat[k0 * 32 + m], Emat[(k0 + 1) * 32 + m]);
    }
  }

  // Init Z = identity in the LDS staging tile (Zh[n][k] = Z[k][n]).
  {
    unsigned int* row = (unsigned int*)&S.Zh[lane][0];
    #pragma unroll
    for (int d = 0; d < 16; ++d) {
      unsigned int val = 0u;
      if (lane == 2 * d)     val = 0x3C00u;      // f16 1.0 in low half
      if (lane == 2 * d + 1) val = 0x3C000000u;  // f16 1.0 in high half
      row[d] = val;
    }
  }
  asm volatile("s_wait_dscnt 0" ::: "memory");

  // B layout: lane holds N=16c+ln; lanes<16 carry K=0..15, lanes>=16 K=16..31,
  // packed ascending -> 2 contiguous b128 loads per fragment.
  FragU bf[2];
  #pragma unroll
  for (int c = 0; c < 2; ++c) {
    const uint4* p = (const uint4*)((const char*)&S.Zh[16 * c + ln][0] + 32 * h);
    bf[c].q[0] = p[0]; bf[c].q[1] = p[1];
  }

  int eacc = 0;
  const int t0 = (k == 0) ? 1 : (CHUNK_LEN * k);
  const int nsteps = (k == 0) ? (CHUNK_LEN - 1) : CHUNK_LEN;
  const float* erow = emis + ((size_t)b * S_LEN + t0) * T_TAGS + lane;

  // One recurrence step: Z <- diag(e_t) * (E^T @ Z), optionally renormalized
  // by 2^-e2 (exponent accumulated), optionally emitting the f32 result.
  auto step = [&](bool renorm, bool emit) {
    float ev = __expf(erow[0]);
    __builtin_prefetch(erow + 4 * T_TAGS, 0, 1);   // global_prefetch_b8
    erow += T_TAGS;
    S.sE[lane] = ev;

    const v8f zero = {};
    v8f d00 = __builtin_amdgcn_wmma_f32_16x16x32_f16(false, af[0].v, false, bf[0].v, (short)0, zero, false, false);
    v8f d01 = __builtin_amdgcn_wmma_f32_16x16x32_f16(false, af[0].v, false, bf[1].v, (short)0, zero, false, false);
    v8f d10 = __builtin_amdgcn_wmma_f32_16x16x32_f16(false, af[1].v, false, bf[0].v, (short)0, zero, false, false);
    v8f d11 = __builtin_amdgcn_wmma_f32_16x16x32_f16(false, af[1].v, false, bf[1].v, (short)0, zero, false, false);

    asm volatile("s_wait_dscnt 0" ::: "memory");
    // e_t values for the row indices this lane will scale: m = 16r + 8h + v.
    float er[2][8];
    #pragma unroll
    for (int r = 0; r < 2; ++r) {
      const float4* ep = (const float4*)&S.sE[16 * r + 8 * h];
      float4 e0 = ep[0], e1 = ep[1];
      er[r][0] = e0.x; er[r][1] = e0.y; er[r][2] = e0.z; er[r][3] = e0.w;
      er[r][4] = e1.x; er[r][5] = e1.y; er[r][6] = e1.z; er[r][7] = e1.w;
    }

    float sc = 1.0f;
    if (renorm) {
      float gm = 0.0f;
      #pragma unroll
      for (int v = 0; v < 8; ++v)
        gm = fmaxf(gm, fmaxf(fmaxf(d00[v], d01[v]), fmaxf(d10[v], d11[v])));
      #pragma unroll
      for (int off = 16; off >= 1; off >>= 1)
        gm = fmaxf(gm, __shfl_xor(gm, off, 32));
      // gm = f * 2^e2, f in [0.5,1); scale by exact power of two 2^-e2.
      const int e2 = (int)((__float_as_uint(gm) >> 23) & 0xFF) - 126;
      sc = __int_as_float((127 - e2) << 23);
      eacc += e2;
    }

    // Scale rows by e_t (and sc), convert to f16, restage to LDS in B layout.
    // D layout: VGPR v holds local M = v + 8h; lane holds local N = ln.
    #pragma unroll
    for (int r = 0; r < 2; ++r) {
      #pragma unroll
      for (int c = 0; c < 2; ++c) {
        const v8f& D = (r == 0) ? (c == 0 ? d00 : d01) : (c == 0 ? d10 : d11);
        float xs[8];
        #pragma unroll
        for (int v = 0; v < 8; ++v) xs[v] = D[v] * er[r][v] * sc;
        uint4 q;
        q.x = pack_f16(xs[0], xs[1]);
        q.y = pack_f16(xs[2], xs[3]);
        q.z = pack_f16(xs[4], xs[5]);
        q.w = pack_f16(xs[6], xs[7]);
        const int n = 16 * c + ln;
        *(uint4*)((char*)&S.Zh[n][0] + (32 * r + 16 * h)) = q;
        if (emit) {   // emit chunk matrix M_k^T (f32, row-major), last step only
          float* out = mats + (size_t)worker * 1024 + (16 * r + 8 * h) * 32 + n;
          #pragma unroll
          for (int v = 0; v < 8; ++v) out[v * 32] = xs[v];
        }
      }
    }
    asm volatile("s_wait_dscnt 0" ::: "memory");
    #pragma unroll
    for (int c = 0; c < 2; ++c) {
      const uint4* p = (const uint4*)((const char*)&S.Zh[16 * c + ln][0] + 32 * h);
      bf[c].q[0] = p[0]; bf[c].q[1] = p[1];
    }
  };

  // Prologue (chunk 0 only): 3 plain steps so remaining count is a multiple of 4.
  if (k == 0) { step(false, false); step(false, false); step(false, false); }

  // Main loop: groups of 4 steps, unconditional renorm on the 4th.
  const int ngroups = ((k == 0) ? (nsteps - 3) : nsteps) >> 2;   // 31 or 32
  for (int g = 0; g < ngroups - 1; ++g) {
    step(false, false); step(false, false); step(false, false); step(true, false);
  }
  // Final group: renorm + emit on the last step, straight-line.
  step(false, false); step(false, false); step(false, false); step(true, true);

  if (lane == 0) exps[worker] = eacc;
}

// ---------------- Stage 2: per-batch combine -> fwd logZ ----------------
__global__ void __launch_bounds__(256)
crf_combine(const float* __restrict__ emis, const float* __restrict__ startT,
            const float* __restrict__ endT, const float* __restrict__ mats,
            const int* __restrict__ exps, float* __restrict__ fwd)
{
  __shared__ float su[8][32];
  const int lane = threadIdx.x & 31;
  const int wid  = threadIdx.x >> 5;
  const int b = blockIdx.x * 8 + wid;

  float a = startT[lane] + emis[(size_t)b * S_LEN * T_TAGS + lane];  // alpha_0
  for (int k = 0; k < N_CHUNK; ++k) {
    float m = a;
    #pragma unroll
    for (int off = 16; off >= 1; off >>= 1) m = fmaxf(m, __shfl_xor(m, off, 32));
    su[wid][lane] = __expf(a - m);
    asm volatile("s_wait_dscnt 0" ::: "memory");
    // w[j] = sum_i u[i] * M_k[i][j] = sum_i u[i] * Mt[j][i]  (Mt row j contiguous)
    const float* R = mats + ((size_t)(b * N_CHUNK + k)) * 1024 + lane * 32;
    float acc = 0.0f;
    #pragma unroll 8
    for (int i = 0; i < 32; ++i) acc = fmaf(su[wid][i], R[i], acc);
    a = logf(acc) + m + (float)exps[b * N_CHUNK + k] * 0.69314718055994530942f;
  }
  a += endT[lane];
  float m2 = a;
  #pragma unroll
  for (int off = 16; off >= 1; off >>= 1) m2 = fmaxf(m2, __shfl_xor(m2, off, 32));
  float s = __expf(a - m2);
  #pragma unroll
  for (int off = 16; off >= 1; off >>= 1) s += __shfl_xor(s, off, 32);
  if (lane == 0) fwd[b] = m2 + logf(s);
}

// ---------------- Stage 3: gold score + mean(fwd - gold) ----------------
__global__ void __launch_bounds__(256)
crf_gold_reduce(const float* __restrict__ emis, const int* __restrict__ tags,
                const unsigned char* __restrict__ mask, const float* __restrict__ trans,
                const float* __restrict__ startT, const float* __restrict__ endT,
                const float* __restrict__ fwd, float* __restrict__ out)
{
  __shared__ float rp[256];
  __shared__ int   rm[256];
  const int b = blockIdx.x;
  const int tid = threadIdx.x;
  float p = 0.0f; int msum = 0;
  for (int t = tid; t < S_LEN; t += 256) {
    const int tg = tags[(size_t)b * S_LEN + t];
    const int mk = mask[(size_t)b * S_LEN + t] ? 1 : 0;
    msum += mk;
    if (t == 0) {
      p += startT[tg] + emis[((size_t)b * S_LEN) * T_TAGS + tg];
    } else if (mk) {
      const int tp = tags[(size_t)b * S_LEN + t - 1];
      p += trans[tp * T_TAGS + tg] + emis[((size_t)b * S_LEN + t) * T_TAGS + tg];
    }
  }
  rp[tid] = p; rm[tid] = msum;
  __syncthreads();
  for (int s = 128; s > 0; s >>= 1) {
    if (tid < s) { rp[tid] += rp[tid + s]; rm[tid] += rm[tid + s]; }
    __syncthreads();
  }
  if (tid == 0) {
    const int last = rm[0] - 1;
    const int ltag = tags[(size_t)b * S_LEN + last];
    const float gold = rp[0] + endT[ltag];
    atomicAdd(out, (fwd[b] - gold) * (1.0f / (float)B_SZ));
  }
}

// ---------------- launcher ----------------
extern "C" void kernel_launch(void* const* d_in, const int* in_sizes, int n_in,
                              void* d_out, int out_size, void* d_ws, size_t ws_size,
                              hipStream_t stream) {
  const float*         emis   = (const float*)d_in[0];
  const int*           tags   = (const int*)d_in[1];
  const unsigned char* mask   = (const unsigned char*)d_in[2];
  const float*         trans  = (const float*)d_in[3];
  const float*         startT = (const float*)d_in[4];
  const float*         endT   = (const float*)d_in[5];
  float*               outp   = (float*)d_out;

  char* ws = (char*)d_ws;
  float* Emat = (float*)ws;                                          // 4 KB
  float* mats = (float*)(ws + 4096);                                 // 8192*1024*4 = 32 MB
  int*   exps = (int*)(ws + 4096 + (size_t)8192 * 1024 * 4);         // 32 KB
  float* fwd  = (float*)(ws + 4096 + (size_t)8192 * 1024 * 4 + 8192 * 4); // 2 KB

  (void)hipMemsetAsync(d_out, 0, sizeof(float), stream);
  crf_prep_expT<<<4, 256, 0, stream>>>(trans, Emat);
  crf_chunk_scan<<<(B_SZ * N_CHUNK) / 8, 256, 0, stream>>>(emis, Emat, mats, exps);
  crf_combine<<<B_SZ / 8, 256, 0, stream>>>(emis, startT, endT, mats, exps, fwd);
  crf_gold_reduce<<<B_SZ, 256, 0, stream>>>(emis, tags, mask, trans, startT, endT, fwd, outp);
}
